// LPKT_5677946766128
// MI455X (gfx1250) — compile-verified, matrix operand order checked by reading, not run on hardware
//
#include <hip/hip_runtime.h>
#include <hip/hip_bf16.h>
#include <math.h>

// LPKT: B=32, L=100, S=501, DK=64, DE=128.
// Hoist all non-recurrent GEMMs into parallel WMMA kernels; run the 99-step
// recurrence as 32 workgroups (one per batch) with the (501x64) hidden state
// resident in LDS (stride-66 padded against bank conflicts) and the per-step
// (501,64)@(64,64) gate GEMM done with v_wmma_f32_16x16x32_f16.

typedef __attribute__((ext_vector_type(16))) _Float16 v16h;
typedef __attribute__((ext_vector_type(8)))  float    v8f;

#define NB   32
#define LSEQ 100
#define TSEQ 99          // L-1 scan steps
#define SDIM 501
#define SPAD 512
#define DK   64
#define HSTR 66          // padded LDS row stride for h state (bank-conflict free)

__device__ __forceinline__ float sigf(float x) { return 1.f / (1.f + __expf(-x)); }

// K index base for half2 slot j (0..7) and lane group g (lane>>4), per the
// CDNA5 16-bit A-matrix 16x32 layout (B assumed symmetric with N=lane&15).
__device__ __forceinline__ int kmap(int j, int g) {
  return ((j & 4) ? 16 : 0) + g * 8 + 2 * (j & 3);
}

// ---------------------------------------------------------------------------
// Pack X1 = [e_emb | at_emb | a_rep]  (3200 x 256) in f16
// ---------------------------------------------------------------------------
__global__ void __launch_bounds__(256)
pack_x1(const int* __restrict__ e_data, const int* __restrict__ at_data,
        const int* __restrict__ a_data, const float* __restrict__ e_embed,
        const float* __restrict__ at_embed, _Float16* __restrict__ X1) {
  int idx = blockIdx.x * 256 + threadIdx.x;
  if (idx >= NB * LSEQ * 256) return;
  int row = idx >> 8, col = idx & 255;
  float v;
  if (col < 128)      v = e_embed[(size_t)e_data[row] * 128 + col];
  else if (col < 192) v = at_embed[(size_t)at_data[row] * 64 + (col - 128)];
  else                v = (float)a_data[row];
  X1[idx] = (_Float16)v;
}

// ---------------------------------------------------------------------------
// Pack X2 = [learning_pre | it_emb | learn_t] (3168 x 192) and
//      Xe = e_emb[e_next]                      (3168 x 128), f16
// ---------------------------------------------------------------------------
__global__ void __launch_bounds__(256)
pack_x2(const int* __restrict__ e_data, const int* __restrict__ it_data,
        const float* __restrict__ e_embed, const float* __restrict__ it_embed,
        const float* __restrict__ AL, _Float16* __restrict__ X2,
        _Float16* __restrict__ Xe) {
  const int N2 = NB * TSEQ * 192;
  const int NE2 = NB * TSEQ * 128;
  int idx = blockIdx.x * 256 + threadIdx.x;
  if (idx < N2) {
    int row = idx / 192, col = idx - row * 192;
    int b = row / TSEQ, t = row - b * TSEQ;
    float v;
    if (col < 64)       v = (t == 0) ? 0.f : AL[((size_t)b * LSEQ + t - 1) * DK + col];
    else if (col < 128) v = it_embed[(size_t)it_data[b * LSEQ + t] * 64 + (col - 64)];
    else                v = AL[((size_t)b * LSEQ + t) * DK + (col - 128)];
    X2[idx] = (_Float16)v;
  } else if (idx < N2 + NE2) {
    int k = idx - N2;
    int row = k >> 7, col = k & 127;
    int b = row / TSEQ, t = row - b * TSEQ;
    Xe[k] = (_Float16)e_embed[(size_t)e_data[b * LSEQ + t + 1] * 128 + col];
  }
}

// ---------------------------------------------------------------------------
// Generic C(M,64) = A(M,K)f16 @ W(64, ldw)[:, wcol0:wcol0+K]^T + bias
// One wave per 16-row M-tile, 8 waves per block. K multiple of 32.
// ---------------------------------------------------------------------------
__global__ void __launch_bounds__(256)
gemm_wmma_f16(const _Float16* __restrict__ A, int lda, int M, int K,
              const float* __restrict__ W, int ldw, int wcol0,
              const float* __restrict__ bias, float* __restrict__ C) {
  int lane = threadIdx.x & 31;
  int wave = threadIdx.x >> 5;
  int mt = blockIdx.x * 8 + wave;
  if (mt * 16 >= M) return;          // wave-uniform exit; EXEC stays all-ones
  int g = lane >> 4, ln = lane & 15;

  v8f acc[4] = {v8f{}, v8f{}, v8f{}, v8f{}};
  int arow = mt * 16 + ln;
  for (int kt = 0; kt < K; kt += 32) {
    v16h a;
    const _Float16* ap = A + (size_t)arow * lda + kt;
#pragma unroll
    for (int j = 0; j < 8; ++j) {
      int kb = kmap(j, g);
      a[2 * j]     = ap[kb];
      a[2 * j + 1] = ap[kb + 1];
    }
#pragma unroll
    for (int nt = 0; nt < 4; ++nt) {
      v16h bf;
      const float* wp = W + (size_t)(nt * 16 + ln) * ldw + wcol0 + kt;
#pragma unroll
      for (int j = 0; j < 8; ++j) {
        int kb = kmap(j, g);
        bf[2 * j]     = (_Float16)wp[kb];
        bf[2 * j + 1] = (_Float16)wp[kb + 1];
      }
      acc[nt] = __builtin_amdgcn_wmma_f32_16x16x32_f16(
          false, a, false, bf, (short)0, acc[nt], false, false);
    }
  }
#pragma unroll
  for (int nt = 0; nt < 4; ++nt) {
    int col = nt * 16 + ln;
    float bv = bias ? bias[col] : 0.f;
#pragma unroll
    for (int r = 0; r < 8; ++r) {
      int row = mt * 16 + r + 8 * g;
      C[(size_t)row * DK + col] = acc[nt][r] + bv;
    }
  }
}

// ---------------------------------------------------------------------------
// Sequential scan: one workgroup per batch, h-state resident in LDS.
// Gate matrices stored TRANSPOSED ([k][d]) so matvec reads are conflict-free.
// ---------------------------------------------------------------------------
__global__ void __launch_bounds__(256)
lpkt_scan(const int* __restrict__ e_data, const float* __restrict__ q_matrix,
          const float* __restrict__ h0, const float* __restrict__ W2,
          const float* __restrict__ W3, const float* __restrict__ W4,
          const float* __restrict__ W5, const float* __restrict__ G2,
          const float* __restrict__ G3, const float* __restrict__ IT4,
          const float* __restrict__ YE, float* __restrict__ out) {
  extern __shared__ float sm[];
  float* hbuf = sm;                  // 512*66 = 33792
  float* W2hT = hbuf + SPAD * HSTR;  // 64*64 each, [k][d]
  float* W3hT = W2hT + 4096;
  float* W4lgT = W3hT + 4096;
  float* W5hT = W4lgT + 4096;
  float* qe   = W5hT + 4096;         // 512
  float* qn   = qe + SPAD;           // 512
  float* LG   = qn + SPAD;           // 64
  float* fb   = LG + DK;             // 64
  float* ht   = fb + DK;             // 64
  float* part = ht + DK;             // 8*32*4 = 1024
  float* yred = part + 1024;         // 64

  const int b = blockIdx.x;
  const int tid = threadIdx.x;
  const int lane = tid & 31, wave = tid >> 5;
  const int g = lane >> 4, ln = lane & 15;

  // ---- init: h state (padded stride), transposed gate weights ----
  for (int i = tid; i < SPAD * HSTR; i += 256) {
    int s = i / HSTR, c = i - s * HSTR;
    hbuf[i] = (s < SDIM && c < DK) ? h0[s * DK + c] : 0.f;
  }
  for (int i = tid; i < 4096; i += 256) {
    int k = i >> 6, d = i & 63;
    W2hT[i]  = W2[d * 256 + 192 + k];   // W2[:,192:256] (h_tilde block)^T
    W3hT[i]  = W3[d * 256 + 192 + k];
    W4lgT[i] = W4[d * 192 + 64 + k];    // W4[:,64:128]  (LG block)^T
    W5hT[i]  = W5[d * 192 + 128 + k];   // W5[:,128:192] (h_tilde block)^T
  }
  // W4_h WMMA B-fragments (kept in registers for all 99 steps)
  v16h Bf[4][2];
#pragma unroll
  for (int nt = 0; nt < 4; ++nt)
#pragma unroll
    for (int kt = 0; kt < 2; ++kt) {
      const float* wp = W4 + (size_t)(nt * 16 + ln) * 192 + kt * 32;  // W4[:,0:64]
      v16h f;
#pragma unroll
      for (int j = 0; j < 8; ++j) {
        int kb = kmap(j, g);
        f[2 * j]     = (_Float16)wp[kb];
        f[2 * j + 1] = (_Float16)wp[kb + 1];
      }
      Bf[nt][kt] = f;
    }
  __syncthreads();

  // h_tilde0 = q_matrix[e[b,0]] @ h0
  if (tid < DK) {
    const float* qrow = q_matrix + (size_t)e_data[b * LSEQ] * SDIM;
    float acc = 0.f;
    for (int s = 0; s < SDIM; ++s) acc += qrow[s] * hbuf[s * HSTR + tid];
    ht[tid] = acc;
  }
  if (tid == 0) out[b * LSEQ] = 0.f;
  __syncthreads();

  for (int t = 0; t < TSEQ; ++t) {
    const int bt = b * TSEQ + t;
    // --- phase A: (0-63) LG gates | (64-127) y_{t-1} | (128-255) q gathers ---
    if (tid < DK) {
      float m2 = 0.f, m3 = 0.f;
      for (int k = 0; k < DK; ++k) {
        float hv = ht[k];
        m2 += W2hT[k * DK + tid] * hv;
        m3 += W3hT[k * DK + tid] * hv;
      }
      float g2 = G2[(size_t)bt * DK + tid] + m2;
      float g3 = G3[(size_t)bt * DK + tid] + m3;
      LG[tid] = sigf(g3) * (tanhf(g2) + 1.f) * 0.5f;
    } else if (tid < 128) {
      if (t > 0) {                       // prediction for step t-1 uses this ht
        int d = tid - 64;
        float m = 0.f;
        for (int k = 0; k < DK; ++k) m += W5hT[k * DK + d] * ht[k];
        yred[d] = sigf(YE[(size_t)(bt - 1) * DK + d] + m);
      }
    } else {
      const float* qerow = q_matrix + (size_t)e_data[b * LSEQ + t] * SDIM;
      const float* qnrow = q_matrix + (size_t)e_data[b * LSEQ + t + 1] * SDIM;
      for (int s = tid - 128; s < SPAD; s += 128) {
        qe[s] = (s < SDIM) ? qerow[s] : 0.f;
        qn[s] = (s < SDIM) ? qnrow[s] : 0.f;
      }
    }
    __syncthreads();
    // --- phase B: fbias | y-reduce | prefetch next q rows ---
    if (tid < DK) {
      float m = 0.f;
      for (int k = 0; k < DK; ++k) m += W4lgT[k * DK + tid] * LG[k];
      fb[tid] = m + IT4[(size_t)bt * DK + tid];
    } else if (tid == 64) {
      if (t > 0) {
        float s = 0.f;
        for (int d = 0; d < DK; ++d) s += yred[d];
        out[b * LSEQ + t] = s * (1.f / 64.f);
      }
    } else if (tid >= 192 && t + 1 < TSEQ) {
      const float* pe = q_matrix + (size_t)e_data[b * LSEQ + t + 1] * SDIM;
      const float* pn = q_matrix + (size_t)e_data[b * LSEQ + t + 2] * SDIM;
      if (tid < 208)      __builtin_prefetch(pe + (tid - 192) * 32, 0, 0);
      else if (tid < 224) __builtin_prefetch(pn + (tid - 208) * 32, 0, 0);
    }
    __syncthreads();
    // --- phase C: WMMA gamma_f, h update, h_tilde partials ---
    float pacc[4] = {0.f, 0.f, 0.f, 0.f};
#pragma unroll
    for (int mi = 0; mi < 4; ++mi) {
      int mt = wave + mi * 8;       // 32 M-tiles over 8 waves
      int row0 = mt * 16;
      v16h a0, a1;
      {
        const float* hp = hbuf + (size_t)(row0 + ln) * HSTR;
#pragma unroll
        for (int j = 0; j < 8; ++j) {
          int kb = kmap(j, g);
          a0[2 * j]     = (_Float16)hp[kb];
          a0[2 * j + 1] = (_Float16)hp[kb + 1];
          a1[2 * j]     = (_Float16)hp[32 + kb];
          a1[2 * j + 1] = (_Float16)hp[32 + kb + 1];
        }
      }
#pragma unroll
      for (int nt = 0; nt < 4; ++nt) {
        v8f acc = {};
        acc = __builtin_amdgcn_wmma_f32_16x16x32_f16(
            false, a0, false, Bf[nt][0], (short)0, acc, false, false);
        acc = __builtin_amdgcn_wmma_f32_16x16x32_f16(
            false, a1, false, Bf[nt][1], (short)0, acc, false, false);
        int col = nt * 16 + ln;
        float lgc = LG[col], fbc = fb[col];
#pragma unroll
        for (int r = 0; r < 8; ++r) {
          int s = row0 + r + 8 * g;
          float gf = sigf(acc[r] + fbc);
          float hnew = qe[s] * lgc + gf * hbuf[s * HSTR + col];
          hbuf[s * HSTR + col] = hnew;      // padded rows stay exactly 0
          pacc[nt] += qn[s] * hnew;
        }
      }
    }
#pragma unroll
    for (int nt = 0; nt < 4; ++nt) part[(wave * 32 + lane) * 4 + nt] = pacc[nt];
    __syncthreads();
    // --- phase D: reduce h_tilde = q_next^T h ---
    if (tid < DK) {
      int c15 = tid & 15, nt = tid >> 4;
      float acc = 0.f;
      for (int w = 0; w < 8; ++w)
        acc += part[(w * 32 + c15) * 4 + nt] + part[(w * 32 + c15 + 16) * 4 + nt];
      ht[tid] = acc;
    }
    __syncthreads();
  }
  // final prediction y_{98}
  if (tid < DK) {
    float m = 0.f;
    for (int k = 0; k < DK; ++k) m += W5hT[k * DK + tid] * ht[k];
    yred[tid] = sigf(YE[(size_t)(b * TSEQ + TSEQ - 1) * DK + tid] + m);
  }
  __syncthreads();
  if (tid == 0) {
    float s = 0.f;
    for (int d = 0; d < DK; ++d) s += yred[d];
    out[b * LSEQ + LSEQ - 1] = s * (1.f / 64.f);
  }
}

// ---------------------------------------------------------------------------
extern "C" void kernel_launch(void* const* d_in, const int* in_sizes, int n_in,
                              void* d_out, int out_size, void* d_ws, size_t ws_size,
                              hipStream_t stream) {
  const int* e_data  = (const int*)d_in[0];
  const int* a_data  = (const int*)d_in[1];
  const int* at_data = (const int*)d_in[2];
  const int* it_data = (const int*)d_in[3];
  const float* q_matrix = (const float*)d_in[4];
  const float* h0       = (const float*)d_in[5];
  const float* e_embed  = (const float*)d_in[6];
  const float* at_embed = (const float*)d_in[7];
  const float* it_embed = (const float*)d_in[8];
  const float* W1 = (const float*)d_in[9];  const float* b1 = (const float*)d_in[10];
  const float* W2 = (const float*)d_in[11]; const float* b2 = (const float*)d_in[12];
  const float* W3 = (const float*)d_in[13]; const float* b3 = (const float*)d_in[14];
  const float* W4 = (const float*)d_in[15]; const float* b4 = (const float*)d_in[16];
  const float* W5 = (const float*)d_in[17]; const float* b5 = (const float*)d_in[18];
  float* out = (float*)d_out;

  char* ws = (char*)d_ws;
  size_t off = 0;
  auto alloc = [&](size_t bytes) -> void* {
    void* p = ws + off;
    off = (off + bytes + 255) & ~(size_t)255;
    return p;
  };
  _Float16* X1  = (_Float16*)alloc((size_t)NB * LSEQ * 256 * 2);
  float*    AL  = (float*)   alloc((size_t)NB * LSEQ * DK * 4);   // all_learning
  _Float16* X2  = (_Float16*)alloc((size_t)NB * TSEQ * 192 * 2);
  _Float16* Xe  = (_Float16*)alloc((size_t)NB * TSEQ * 128 * 2);
  float*    G2v = (float*)   alloc((size_t)NB * TSEQ * DK * 4);
  float*    G3v = (float*)   alloc((size_t)NB * TSEQ * DK * 4);
  float*    IT4v= (float*)   alloc((size_t)NB * TSEQ * DK * 4);
  float*    YEv = (float*)   alloc((size_t)NB * TSEQ * DK * 4);

  // 1) gather + pack inputs for the W1 GEMM
  pack_x1<<<(NB * LSEQ * 256 + 255) / 256, 256, 0, stream>>>(
      e_data, at_data, a_data, e_embed, at_embed, X1);
  // 2) all_learning = X1 @ W1^T + b1   (3200 x 256 x 64)
  gemm_wmma_f16<<<25, 256, 0, stream>>>(X1, 256, NB * LSEQ, 256, W1, 256, 0, b1, AL);
  // 3) pack scan-time inputs
  pack_x2<<<(NB * TSEQ * (192 + 128) + 255) / 256, 256, 0, stream>>>(
      e_data, it_data, e_embed, it_embed, AL, X2, Xe);
  // 4) hoisted gate GEMMs (everything except the h_tilde-dependent block)
  gemm_wmma_f16<<<25, 256, 0, stream>>>(X2, 192, NB * TSEQ, 192, W2, 256, 0,   b2, G2v);
  gemm_wmma_f16<<<25, 256, 0, stream>>>(X2, 192, NB * TSEQ, 192, W3, 256, 0,   b3, G3v);
  gemm_wmma_f16<<<25, 256, 0, stream>>>(X2 + 64, 192, NB * TSEQ, 64, W4, 192, 128, b4, IT4v);
  gemm_wmma_f16<<<25, 256, 0, stream>>>(Xe, 128, NB * TSEQ, 128, W5, 192, 0,   b5, YEv);
  // 5) sequential scan: one WG per batch, h-state in LDS (~210 KB/WG < 320 KB)
  size_t lds_bytes = (size_t)(SPAD * HSTR + 4 * 4096 + 2 * SPAD + 4 * 64 + 1024) * 4;
  lpkt_scan<<<NB, 256, lds_bytes, stream>>>(
      e_data, q_matrix, h0, W2, W3, W4, W5, G2v, G3v, IT4v, YEv, out);
}
